// ParticleNet_86775519248877
// MI455X (gfx1250) — matrix operand order repaired and checked by reading, not compiled, not run on hardware
//
#include <hip/hip_runtime.h>

typedef _Float16 h16;
typedef __attribute__((ext_vector_type(16))) _Float16 v16h;
typedef __attribute__((ext_vector_type(8)))  _Float16 v8h;
typedef __attribute__((ext_vector_type(8)))  float    v8f;

#define GB   128                 // graphs (batch)
#define NN   128                 // nodes per graph
#define KNB  16                  // neighbors
#define NE   (GB*NN*KNB)         // 262144 edges
#define BNEPS 1e-5f
#define KCHUNK 128               // K-depth staged in LDS per iteration

// ---------------------------------------------------------------------------
// Weight convert f32 -> f16 row-major (rowsPad == rows for all but w11).
// ---------------------------------------------------------------------------
__global__ void pn_cvtw(const float* __restrict__ src, h16* __restrict__ dst,
                        int rows, int cols, int rowsPad) {
  size_t total  = (size_t)rowsPad * cols;
  size_t stride = (size_t)gridDim.x * blockDim.x;
  for (size_t i = (size_t)blockIdx.x * blockDim.x + threadIdx.x; i < total; i += stride) {
    int r = (int)(i / cols);
    int c = (int)(i % cols);
    dst[i] = (r < rows) ? (h16)src[(size_t)r * cols + c] : (h16)0.f;
  }
}

// w11 [12,64] -> padded/remapped [32,64]: rows 0-5 = xi ch0-5, rows 8-13 =
// (xj-xi) ch0-5, all other rows zero (matches 8-channel-padded x features).
__global__ void pn_cvtw11(const float* __restrict__ src, h16* __restrict__ dst) {
  int i = blockIdx.x * blockDim.x + threadIdx.x;
  if (i >= 32 * 64) return;
  int r = i >> 6, c = i & 63;
  float v = 0.f;
  if (r < 6)                 v = src[r * 64 + c];
  else if (r >= 8 && r < 14) v = src[(r - 2) * 64 + c];
  dst[i] = (h16)v;
}

// x f32 [B,N,6] -> x0h f16 [B,N,8] zero-padded.
__global__ void pn_padx(const float* __restrict__ x, h16* __restrict__ xh) {
  size_t total  = (size_t)GB * NN * 8;
  size_t stride = (size_t)gridDim.x * blockDim.x;
  for (size_t i = (size_t)blockIdx.x * blockDim.x + threadIdx.x; i < total; i += stride) {
    int    c  = (int)(i & 7);
    size_t bn = i >> 3;
    xh[i] = (c < 6) ? (h16)x[bn * 6 + c] : (h16)0.f;
  }
}

__global__ void pn_zero512(float* __restrict__ p) {
  if (threadIdx.x < 512) p[threadIdx.x] = 0.f;
}

// ---------------------------------------------------------------------------
// kNN in feature space (f16 features, f32 distance accumulate):
// one block per (graph, node), 128 threads (= N); 16 rounds of LDS argmin
// with lower-index tie-break (matches top_k stable order; self dist = 0).
// ---------------------------------------------------------------------------
__global__ void pn_knn(const h16* __restrict__ x, int C, int* __restrict__ idx) {
  int bn = blockIdx.x;
  int b  = bn >> 7;
  int n  = bn & (NN - 1);
  int m  = threadIdx.x;               // candidate node
  const h16* xb = x + (size_t)b * NN * C;
  const h16* xn = xb + (size_t)n * C;
  const h16* xm = xb + (size_t)m * C;
  float d = 0.f;
  for (int c = 0; c < C; ++c) { float t = (float)xn[c] - (float)xm[c]; d += t * t; }
  __shared__ float dist[NN];
  __shared__ float rv[NN];
  __shared__ int   ri[NN];
  dist[m] = d;
  __syncthreads();
  for (int t = 0; t < KNB; ++t) {
    rv[m] = dist[m]; ri[m] = m;
    __syncthreads();
    for (int s = 64; s > 0; s >>= 1) {
      if (m < s) {
        float ov = rv[m + s]; int oi = ri[m + s];
        if (ov < rv[m] || (ov == rv[m] && oi < ri[m])) { rv[m] = ov; ri[m] = oi; }
      }
      __syncthreads();
    }
    if (m == 0) { idx[(size_t)bn * KNB + t] = ri[0]; dist[ri[0]] = 3.4e38f; }
    __syncthreads();
  }
}

// ---------------------------------------------------------------------------
// Cooperative, swizzled weight-slab staging: W[kc:kc+kcl, n0:n0+64] -> LDS in
// WMMA B-fragment order; lane L's 16 fragment halves are contiguous 32B.
// ---------------------------------------------------------------------------
__device__ __forceinline__ void pn_stage_w(h16* __restrict__ wlds,
                                           const h16* __restrict__ W,
                                           int kc, int kcl, int n0, int Nc, int t) {
  for (int e0 = t; e0 < kcl * 64; e0 += 256) {
    int k = e0 >> 6;                  // chunk-local k
    int n = e0 & 63;                  // slab-local n (coalesced global reads)
    h16 v = W[(size_t)(kc + k) * Nc + (n0 + n)];
    int kb = k >> 5;
    int L  = (n & 15) | (((k >> 3) & 1) << 4);
    int e  = (k & 7) | (((k >> 4) & 1) << 3);
    wlds[((((kb << 2) + (n >> 4)) << 5) + L) * 16 + e] = v;
  }
}

// ---------------------------------------------------------------------------
// Edge A chunk, fully branch-free: region selection is done on addresses
// (v_cndmask), then two unconditional 16B loads and a packed f16 subtract.
// zb is a 16B-aligned, zero-filled scratch backing the xi/zero-pad regions.
// ---------------------------------------------------------------------------
__device__ __forceinline__ v8h pn_edge_chunk(const h16* __restrict__ xi,
                                             const h16* __restrict__ xj,
                                             const h16* __restrict__ zb,
                                             int cb, int CPAD, int twoC) {
  bool inXi  = cb < CPAD;
  bool inDif = (cb >= CPAD) && (cb < twoC);
  const h16* pa = inXi ? (xi + cb) : (inDif ? (xj + (cb - CPAD)) : zb);
  const h16* pb = inDif ? (xi + (cb - CPAD)) : zb;
  v8h a = *(const v8h*)pa;
  v8h b = *(const v8h*)pb;
  return a - b;                       // v_pk_add_f16 with neg modifiers
}

// ---------------------------------------------------------------------------
// Edge GEMM: rows are edges; A = [xi, xj-xi] built on the fly from f16 node
// features (no E x 2C staging). f16 WMMA, f32 accumulate. Each wave: 16 rows
// x 64 cols (4 tiles); block = 8 waves = 128 rows. B fragments from LDS slab.
// ---------------------------------------------------------------------------
__global__ void pn_gemm_edge(const h16* __restrict__ xh, const int* __restrict__ idx,
                             const h16* __restrict__ W, const float* __restrict__ bias,
                             const h16* __restrict__ zb,
                             h16* __restrict__ Y, int CPAD, int Kpad, int Nc) {
  __shared__ h16 wlds[(KCHUNK / 32) * 4 * 32 * 16];     // 16 KB
  const int t     = threadIdx.x;
  const int lane  = t & 31;
  const int wave  = t >> 5;
  const int mlane = lane & 15;
  const int hi    = lane >> 4;
  const int row0  = blockIdx.x * 128 + wave * 16;
  const int n0    = blockIdx.y * 64;
  const int r     = row0 + mlane;                 // this lane's edge id
  const int b     = r >> 11;                      // / (N*K) = 2048
  const int n     = (r >> 4) & (NN - 1);
  const int j     = idx[r];
  const int twoC  = 2 * CPAD;
  const h16* xi = xh + ((size_t)b * NN + n) * CPAD;
  const h16* xj = xh + ((size_t)b * NN + j) * CPAD;

  v8f acc[4] = {};
  for (int kc = 0; kc < Kpad; kc += KCHUNK) {
    const int kcl = (Kpad - kc < KCHUNK) ? (Kpad - kc) : KCHUNK;
    pn_stage_w(wlds, W, kc, kcl, n0, Nc, t);
    __syncthreads();
    for (int kb = 0; kb < kcl; kb += 32) {
      // ISA A layout: lanes<16 hold K kb+0..7 / kb+16..23, lanes>=16 hold +8.
      v8h lo  = pn_edge_chunk(xi, xj, zb, kc + kb + hi * 8,      CPAD, twoC);
      v8h hi8 = pn_edge_chunk(xi, xj, zb, kc + kb + 16 + hi * 8, CPAD, twoC);
      v16h afrag;
#pragma unroll
      for (int e = 0; e < 8; ++e) { afrag[e] = lo[e]; afrag[8 + e] = hi8[e]; }
      const h16* lbase = wlds + (((kb >> 5) << 2) << 5) * 16 + lane * 16;
#pragma unroll
      for (int nt = 0; nt < 4; ++nt) {
        v16h bfrag = *(const v16h*)(lbase + (nt << 5) * 16);
        acc[nt] = __builtin_amdgcn_wmma_f32_16x16x32_f16(false, afrag, false, bfrag,
                                                         (short)0, acc[nt], false, false);
      }
    }
    __syncthreads();
  }
#pragma unroll
  for (int nt = 0; nt < 4; ++nt) {
    const int   nn2 = n0 + nt * 16 + mlane;
    const float bv  = bias[nn2];
    h16* yp = Y + (size_t)(row0 + hi * 8) * Nc + nn2;
#pragma unroll
    for (int v = 0; v < 8; ++v) yp[(size_t)v * Nc] = (h16)(acc[nt][v] + bv);
  }
}

// ---------------------------------------------------------------------------
// Dense GEMM: A f16 [M,Ka] row-major, W f16 [Ka,Nc] row-major, Y f16 [M,Nc].
// A fragments: two 16B v8h global loads per lane; B fragments from LDS slab.
// ---------------------------------------------------------------------------
__global__ void pn_gemm(const h16* __restrict__ A, const h16* __restrict__ W,
                        const float* __restrict__ bias, h16* __restrict__ Y,
                        int Ka, int Nc) {
  __shared__ h16 wlds[(KCHUNK / 32) * 4 * 32 * 16];     // 16 KB
  const int t     = threadIdx.x;
  const int lane  = t & 31;
  const int wave  = t >> 5;
  const int mlane = lane & 15;
  const int hi    = lane >> 4;
  const int row0  = blockIdx.x * 128 + wave * 16;
  const int n0    = blockIdx.y * 64;
  const h16* arow = A + (size_t)(row0 + mlane) * Ka;

  v8f acc[4] = {};
  for (int kc = 0; kc < Ka; kc += KCHUNK) {
    const int kcl = (Ka - kc < KCHUNK) ? (Ka - kc) : KCHUNK;
    pn_stage_w(wlds, W, kc, kcl, n0, Nc, t);
    __syncthreads();
    for (int kb = 0; kb < kcl; kb += 32) {
      v8h alo = *(const v8h*)(arow + kc + kb + hi * 8);
      v8h ahi = *(const v8h*)(arow + kc + kb + 16 + hi * 8);
      v16h afrag;
#pragma unroll
      for (int e = 0; e < 8; ++e) { afrag[e] = alo[e]; afrag[8 + e] = ahi[e]; }
      const h16* lbase = wlds + (((kb >> 5) << 2) << 5) * 16 + lane * 16;
#pragma unroll
      for (int nt = 0; nt < 4; ++nt) {
        v16h bfrag = *(const v16h*)(lbase + (nt << 5) * 16);
        acc[nt] = __builtin_amdgcn_wmma_f32_16x16x32_f16(false, afrag, false, bfrag,
                                                         (short)0, acc[nt], false, false);
      }
    }
    __syncthreads();
  }
#pragma unroll
  for (int nt = 0; nt < 4; ++nt) {
    const int   nn2 = n0 + nt * 16 + mlane;
    const float bv  = bias[nn2];
    h16* yp = Y + (size_t)(row0 + hi * 8) * Nc + nn2;
#pragma unroll
    for (int v = 0; v < 8; ++v) yp[(size_t)v * Nc] = (h16)(acc[nt][v] + bv);
  }
}

// ---------------------------------------------------------------------------
// Per-channel sum / sum-of-squares over all rows (BatchNorm training stats).
// LDS-binned, one global atomic per channel per block.
// ---------------------------------------------------------------------------
__global__ void pn_stats(const h16* __restrict__ h, size_t total, int cmask,
                         float* __restrict__ stats) {
  __shared__ float ss[256];
  __shared__ float sq[256];
  int t = threadIdx.x;
  ss[t] = 0.f; sq[t] = 0.f;
  __syncthreads();
  size_t stride = (size_t)gridDim.x * blockDim.x;
  for (size_t i = (size_t)blockIdx.x * blockDim.x + t; i < total; i += stride) {
    float v = (float)h[i];
    int c = (int)(i & (size_t)cmask);
    atomicAdd(&ss[c], v);
    atomicAdd(&sq[c], v * v);
  }
  __syncthreads();
  if (t <= cmask) {
    atomicAdd(&stats[t],       ss[t]);
    atomicAdd(&stats[256 + t], sq[t]);
  }
}

// BN (train) + ReLU, f16 -> f16 (feeds next GEMM's A operand).
__global__ void pn_bnrelu(const h16* __restrict__ h, const float* __restrict__ stats,
                          const float* __restrict__ g, const float* __restrict__ beta,
                          h16* __restrict__ out, size_t rowsM, int cmask) {
  size_t total  = rowsM * (size_t)(cmask + 1);
  float  invM   = 1.f / (float)rowsM;
  size_t stride = (size_t)gridDim.x * blockDim.x;
  for (size_t i = (size_t)blockIdx.x * blockDim.x + threadIdx.x; i < total; i += stride) {
    int   c    = (int)(i & (size_t)cmask);
    float mean = stats[c] * invM;
    float var  = stats[256 + c] * invM - mean * mean;
    float sc   = rsqrtf(var + BNEPS) * g[c];
    float v    = ((float)h[i] - mean) * sc + beta[c];
    out[i] = (h16)fmaxf(v, 0.f);
  }
}

// BN (train) + ReLU + max over K neighbors -> next node features (f16).
__global__ void pn_maxagg(const h16* __restrict__ h, const float* __restrict__ stats,
                          const float* __restrict__ g, const float* __restrict__ beta,
                          h16* __restrict__ xout, int C, int cshift) {
  size_t total  = (size_t)GB * NN * C;
  float  invM   = 1.f / (float)NE;
  size_t stride = (size_t)gridDim.x * blockDim.x;
  int    cmask  = C - 1;
  for (size_t i = (size_t)blockIdx.x * blockDim.x + threadIdx.x; i < total; i += stride) {
    int    c   = (int)(i & (size_t)cmask);
    size_t bn  = i >> cshift;
    float mean = stats[c] * invM;
    float var  = stats[256 + c] * invM - mean * mean;
    float sc   = rsqrtf(var + BNEPS) * g[c];
    float sh   = beta[c] - mean * sc;
    const h16* hp = h + (bn * KNB) * (size_t)C + c;
    float mx = -3.4e38f;
    for (int kk = 0; kk < KNB; ++kk)
      mx = fmaxf(mx, (float)hp[(size_t)kk * C] * sc + sh);
    xout[i] = (h16)fmaxf(mx, 0.f);          // relu(max) == max(relu)
  }
}

// Global mean + max pooling over nodes -> pooled[b,0:256]=mean,[256:512]=max.
__global__ void pn_pool(const h16* __restrict__ x3, h16* __restrict__ pooled) {
  int b = blockIdx.x;
  int c = threadIdx.x;                      // 256 channels
  const h16* xp = x3 + ((size_t)b * NN) * 256 + c;
  float s = 0.f, mx = -3.4e38f;
  for (int nn2 = 0; nn2 < NN; ++nn2) { float v = (float)xp[(size_t)nn2 * 256]; s += v; mx = fmaxf(mx, v); }
  pooled[(size_t)b * 512 + c]       = (h16)(s * (1.f / NN));
  pooled[(size_t)b * 512 + 256 + c] = (h16)mx;
}

// Head BN (stats over the 128 batch rows) + ReLU, one thread per channel.
__global__ void pn_bnsmall(const h16* __restrict__ y, const float* __restrict__ g,
                           const float* __restrict__ beta, h16* __restrict__ z, int Cc) {
  int c = blockIdx.x * blockDim.x + threadIdx.x;
  if (c >= Cc) return;
  float s = 0.f, sq = 0.f;
  for (int rr = 0; rr < GB; ++rr) { float v = (float)y[(size_t)rr * Cc + c]; s += v; sq += v * v; }
  float mean = s * (1.f / GB);
  float var  = sq * (1.f / GB) - mean * mean;
  float sc   = rsqrtf(var + BNEPS) * g[c];
  float sh   = beta[c] - mean * sc;
  for (int rr = 0; rr < GB; ++rr) {
    float v = (float)y[(size_t)rr * Cc + c] * sc + sh;
    z[(size_t)rr * Cc + c] = (h16)fmaxf(v, 0.f);
  }
}

// Final 128x128 @ 128x2 + bias -> f32 output [128,2].
__global__ void pn_fc3(const h16* __restrict__ z, const float* __restrict__ w,
                       const float* __restrict__ bias, float* __restrict__ out) {
  int t = threadIdx.x;
  if (t >= 256) return;
  int b = t >> 1, o = t & 1;
  float acc = 0.f;
  for (int c = 0; c < 128; ++c) acc += (float)z[(size_t)b * 128 + c] * w[(size_t)c * 2 + o];
  out[t] = acc + bias[o];
}

// ---------------------------------------------------------------------------
static inline void* pn_alloc(char*& p, size_t bytes) {
  void* r = (void*)p;
  p += (bytes + 255) & ~(size_t)255;
  return r;
}

extern "C" void kernel_launch(void* const* d_in, const int* in_sizes, int n_in,
                              void* d_out, int out_size, void* d_ws, size_t ws_size,
                              hipStream_t stream) {
  (void)in_sizes; (void)n_in; (void)out_size; (void)ws_size;
  const float* x   = (const float*)d_in[0];
  const float* w11 = (const float*)d_in[1];  const float* b11 = (const float*)d_in[2];
  const float* g11 = (const float*)d_in[3];  const float* e11 = (const float*)d_in[4];
  const float* w12 = (const float*)d_in[5];  const float* b12 = (const float*)d_in[6];
  const float* g12 = (const float*)d_in[7];  const float* e12 = (const float*)d_in[8];
  const float* w21 = (const float*)d_in[9];  const float* b21 = (const float*)d_in[10];
  const float* g21 = (const float*)d_in[11]; const float* e21 = (const float*)d_in[12];
  const float* w22 = (const float*)d_in[13]; const float* b22 = (const float*)d_in[14];
  const float* g22 = (const float*)d_in[15]; const float* e22 = (const float*)d_in[16];
  const float* w31 = (const float*)d_in[17]; const float* b31 = (const float*)d_in[18];
  const float* g31 = (const float*)d_in[19]; const float* e31 = (const float*)d_in[20];
  const float* w32 = (const float*)d_in[21]; const float* b32 = (const float*)d_in[22];
  const float* g32 = (const float*)d_in[23]; const float* e32 = (const float*)d_in[24];
  const float* fw1 = (const float*)d_in[25]; const float* fb1 = (const float*)d_in[26];
  const float* fg1 = (const float*)d_in[27]; const float* fe1 = (const float*)d_in[28];
  const float* fw2 = (const float*)d_in[29]; const float* fb2 = (const float*)d_in[30];
  const float* fg2 = (const float*)d_in[31]; const float* fe2 = (const float*)d_in[32];
  const float* fw3 = (const float*)d_in[33]; const float* fb3 = (const float*)d_in[34];
  float* out = (float*)d_out;

  // ---- workspace layout ----
  char* p = (char*)d_ws;
  h16*  w11h = (h16*)pn_alloc(p, (size_t)32 * 64 * 2);
  h16*  w12h = (h16*)pn_alloc(p, (size_t)64 * 64 * 2);
  h16*  w21h = (h16*)pn_alloc(p, (size_t)128 * 128 * 2);
  h16*  w22h = (h16*)pn_alloc(p, (size_t)128 * 128 * 2);
  h16*  w31h = (h16*)pn_alloc(p, (size_t)256 * 256 * 2);
  h16*  w32h = (h16*)pn_alloc(p, (size_t)256 * 256 * 2);
  h16*  fw1h = (h16*)pn_alloc(p, (size_t)512 * 256 * 2);
  h16*  fw2h = (h16*)pn_alloc(p, (size_t)256 * 128 * 2);
  int*  idx  = (int*)pn_alloc(p, (size_t)NE * 4);
  float* st  = (float*)pn_alloc(p, 512 * 4);
  float* zbf = (float*)pn_alloc(p, 512 * 4);            // zero scratch (pad region)
  h16*  x0h  = (h16*)pn_alloc(p, (size_t)GB * NN * 8   * 2);
  h16*  x1h  = (h16*)pn_alloc(p, (size_t)GB * NN * 64  * 2);
  h16*  x2h  = (h16*)pn_alloc(p, (size_t)GB * NN * 128 * 2);
  h16*  x3h  = (h16*)pn_alloc(p, (size_t)GB * NN * 256 * 2);
  h16*  pooled = (h16*)pn_alloc(p, (size_t)GB * 512 * 2);
  h16*  y1 = (h16*)pn_alloc(p, (size_t)GB * 256 * 2);
  h16*  z1 = (h16*)pn_alloc(p, (size_t)GB * 256 * 2);
  h16*  y2 = (h16*)pn_alloc(p, (size_t)GB * 128 * 2);
  h16*  z2 = (h16*)pn_alloc(p, (size_t)GB * 128 * 2);
  h16*  Hb = (h16*)pn_alloc(p, (size_t)NE * 256 * 2);   // pre-BN GEMM output
  h16*  Ab = (h16*)pn_alloc(p, (size_t)NE * 256 * 2);   // post-BN GEMM input
  const h16* zb = (const h16*)zbf;

  const dim3 blk256(256);
  // ---- weight conversion (L2-resident thereafter) + zero scratch ----
  pn_cvtw11<<<dim3(8), blk256, 0, stream>>>(w11, w11h);
  pn_cvtw<<<dim3(16),  blk256, 0, stream>>>(w12, w12h, 64, 64, 64);
  pn_cvtw<<<dim3(64),  blk256, 0, stream>>>(w21, w21h, 128, 128, 128);
  pn_cvtw<<<dim3(64),  blk256, 0, stream>>>(w22, w22h, 128, 128, 128);
  pn_cvtw<<<dim3(256), blk256, 0, stream>>>(w31, w31h, 256, 256, 256);
  pn_cvtw<<<dim3(256), blk256, 0, stream>>>(w32, w32h, 256, 256, 256);
  pn_cvtw<<<dim3(512), blk256, 0, stream>>>(fw1, fw1h, 512, 256, 512);
  pn_cvtw<<<dim3(128), blk256, 0, stream>>>(fw2, fw2h, 256, 128, 256);
  pn_padx<<<dim3(512), blk256, 0, stream>>>(x, x0h);
  pn_zero512<<<1, 512, 0, stream>>>(zbf);

  const dim3 gK(GB * NN), bK(128);
  const dim3 gS(2048), gE1(2048, 1), gE2(2048, 2), gE4(2048, 4);

  // ================= EdgeConv 1 (C=6 padded to 8 -> 64) =================
  pn_knn<<<gK, bK, 0, stream>>>(x0h, 8, idx);
  pn_gemm_edge<<<gE1, blk256, 0, stream>>>(x0h, idx, w11h, b11, zb, Hb, 8, 32, 64);
  pn_zero512<<<1, 512, 0, stream>>>(st);
  pn_stats<<<gS, blk256, 0, stream>>>(Hb, (size_t)NE * 64, 63, st);
  pn_bnrelu<<<dim3(4096), blk256, 0, stream>>>(Hb, st, g11, e11, Ab, (size_t)NE, 63);
  pn_gemm<<<gE1, blk256, 0, stream>>>(Ab, w12h, b12, Hb, 64, 64);
  pn_zero512<<<1, 512, 0, stream>>>(st);
  pn_stats<<<gS, blk256, 0, stream>>>(Hb, (size_t)NE * 64, 63, st);
  pn_maxagg<<<gS, blk256, 0, stream>>>(Hb, st, g12, e12, x1h, 64, 6);

  // ================= EdgeConv 2 (64 -> 128) =================
  pn_knn<<<gK, bK, 0, stream>>>(x1h, 64, idx);
  pn_gemm_edge<<<gE2, blk256, 0, stream>>>(x1h, idx, w21h, b21, zb, Hb, 64, 128, 128);
  pn_zero512<<<1, 512, 0, stream>>>(st);
  pn_stats<<<gS, blk256, 0, stream>>>(Hb, (size_t)NE * 128, 127, st);
  pn_bnrelu<<<dim3(4096), blk256, 0, stream>>>(Hb, st, g21, e21, Ab, (size_t)NE, 127);
  pn_gemm<<<gE2, blk256, 0, stream>>>(Ab, w22h, b22, Hb, 128, 128);
  pn_zero512<<<1, 512, 0, stream>>>(st);
  pn_stats<<<gS, blk256, 0, stream>>>(Hb, (size_t)NE * 128, 127, st);
  pn_maxagg<<<gS, blk256, 0, stream>>>(Hb, st, g22, e22, x2h, 128, 7);

  // ================= EdgeConv 3 (128 -> 256) =================
  pn_knn<<<gK, bK, 0, stream>>>(x2h, 128, idx);
  pn_gemm_edge<<<gE4, blk256, 0, stream>>>(x2h, idx, w31h, b31, zb, Hb, 128, 256, 256);
  pn_zero512<<<1, 512, 0, stream>>>(st);
  pn_stats<<<gS, blk256, 0, stream>>>(Hb, (size_t)NE * 256, 255, st);
  pn_bnrelu<<<dim3(4096), blk256, 0, stream>>>(Hb, st, g31, e31, Ab, (size_t)NE, 255);
  pn_gemm<<<gE4, blk256, 0, stream>>>(Ab, w32h, b32, Hb, 256, 256);
  pn_zero512<<<1, 512, 0, stream>>>(st);
  pn_stats<<<gS, blk256, 0, stream>>>(Hb, (size_t)NE * 256, 255, st);
  pn_maxagg<<<gS, blk256, 0, stream>>>(Hb, st, g32, e32, x3h, 256, 8);

  // ================= Head =================
  pn_pool<<<dim3(GB), blk256, 0, stream>>>(x3h, pooled);
  pn_gemm<<<dim3(1, 4), blk256, 0, stream>>>(pooled, fw1h, fb1, y1, 512, 256);
  pn_bnsmall<<<1, 256, 0, stream>>>(y1, fg1, fe1, z1, 256);
  pn_gemm<<<dim3(1, 2), blk256, 0, stream>>>(z1, fw2h, fb2, y2, 256, 128);
  pn_bnsmall<<<1, 256, 0, stream>>>(y2, fg2, fe2, z2, 128);
  pn_fc3<<<1, 256, 0, stream>>>(z2, fw3, fb3, out);
}
